// MemoryBlock_36953898615487
// MI455X (gfx1250) — matrix operand order repaired and checked by reading, not compiled, not run on hardware
//
#include <hip/hip_runtime.h>

// ---------------------------------------------------------------------------
// MI455X (gfx1250): bf16 WMMA GEMM chain with ASYNC global->LDS staging
// ---------------------------------------------------------------------------

typedef __bf16 bf16;
typedef __attribute__((ext_vector_type(8)))  __bf16 v8bf;
typedef __attribute__((ext_vector_type(16))) __bf16 v16bf;
typedef __attribute__((ext_vector_type(8)))  float  v8f;

#define H_DIM 1024
#define S_LEN 4096
#define B_SZ  2
#define TOK   (B_SZ * S_LEN)            // 8192
#define SLOTS 10
#define NKV   (B_SZ * (S_LEN + SLOTS))  // 8212 valid kv rows
#define NPAD  8224                      // padded to multiple of 32
#define FF_DIM 4096

#define LDS_K 48                        // padded LDS row stride (elements)

enum { EPI_NONE = 0, EPI_SILU = 1, EPI_DIFF = 2, EPI_WNEW = 3, EPI_MO = 4 };

struct GemmP {
  const bf16* A;        // I x K, row stride K
  const bf16* B;        // J x K, row stride K (weights)
  const float* bias;    // [J] or nullptr
  float* outF;          // I x J or nullptr
  bf16*  outB;          // I x J or nullptr
  const float* aux;     // EPI_DIFF: kp_f32 ; EPI_WNEW: mem_W f32
  const float* fgate;
  const float* lrate;
  float scale;
  int I, J, K, Nvalid;
  float* moF;           // EPI_MO: compact 8192 x 1024 f32
  bf16*  agate;         // EPI_MO: 8192 x 2048 bf16 (right half)
};

static __device__ __forceinline__ v16bf mk16(v8bf lo, v8bf hi) {
  v16bf r;
#pragma unroll
  for (int i = 0; i < 8; ++i) { r[i] = lo[i]; r[8 + i] = hi[i]; }
  return r;
}

// Issue one double-buffer stage of A/B tiles as async DMA into LDS.
// 4 x global_load_async_to_lds_b128 per thread (ASYNCcnt += 4).
// IOFFSET applies to both the global and the LDS address, which matches the
// mirrored chunk layout (col .. col+15 elements of a 32-element k-slice).
static __device__ __forceinline__ void issue_async(const GemmP& p, int ib,
                                                   int jb, int k0, int t,
                                                   bf16* As, bf16* Bs) {
  int row = t >> 1;
  int col = (t & 1) * 16;
  int gi = ib * 128 + row;
  if (gi >= p.I) gi = 0;  // clamp: those C rows are never stored
  unsigned long long ga =
      (unsigned long long)(uintptr_t)(p.A + (size_t)gi * p.K + k0 + col);
  unsigned la = (unsigned)(uintptr_t)(As + row * LDS_K + col);
  asm volatile("global_load_async_to_lds_b128 %0, %1, off"
               :: "v"(la), "v"(ga) : "memory");
  asm volatile("global_load_async_to_lds_b128 %0, %1, off offset:16"
               :: "v"(la), "v"(ga) : "memory");
  int gj = jb * 128 + row;  // J is a multiple of 128 -> always in range
  unsigned long long gb =
      (unsigned long long)(uintptr_t)(p.B + (size_t)gj * p.K + k0 + col);
  unsigned lb = (unsigned)(uintptr_t)(Bs + row * LDS_K + col);
  asm volatile("global_load_async_to_lds_b128 %0, %1, off"
               :: "v"(lb), "v"(gb) : "memory");
  asm volatile("global_load_async_to_lds_b128 %0, %1, off offset:16"
               :: "v"(lb), "v"(gb) : "memory");
}

static __device__ __forceinline__ void mma_step(const bf16* As, const bf16* Bs,
                                                int wi, int wj, int lane,
                                                v8f (&acc)[2][4]) {
  int l15 = lane & 15, lh = lane >> 4;
  v16bf a[2], b[4];
#pragma unroll
  for (int it = 0; it < 2; ++it) {
    const bf16* ar = As + (wi * 32 + it * 16 + l15) * LDS_K;
    v8bf lo = *(const v8bf*)(ar + lh * 16);
    v8bf hi = *(const v8bf*)(ar + lh * 16 + 8);
    a[it] = mk16(lo, hi);
  }
#pragma unroll
  for (int jt = 0; jt < 4; ++jt) {
    const bf16* br = Bs + (wj * 64 + jt * 16 + l15) * LDS_K + lh * 16;
    b[jt] = mk16(*(const v8bf*)(br), *(const v8bf*)(br + 8));
  }
#pragma unroll
  for (int it = 0; it < 2; ++it)
#pragma unroll
    for (int jt = 0; jt < 4; ++jt)
      acc[it][jt] = __builtin_amdgcn_wmma_f32_16x16x32_bf16(
          false, a[it], false, b[jt], (short)0, acc[it][jt], false, false);
}

template <int EPI>
__global__ __launch_bounds__(256) void gemm_k(GemmP p) {
  __shared__ bf16 As[2][128 * LDS_K];
  __shared__ bf16 Bs[2][128 * LDS_K];
  int t = threadIdx.x;
  int ib = blockIdx.y, jb = blockIdx.x;
  int wave = t >> 5, lane = t & 31;
  int wi = wave & 3, wj = wave >> 2;

  v8f acc[2][4];
#pragma unroll
  for (int it = 0; it < 2; ++it)
#pragma unroll
    for (int jt = 0; jt < 4; ++jt)
#pragma unroll
      for (int v = 0; v < 8; ++v) acc[it][jt][v] = 0.0f;

  int nk = p.K >> 5;
  issue_async(p, ib, jb, 0, t, As[0], Bs[0]);
  for (int kb = 0; kb < nk; ++kb) {
    int cur = kb & 1;
    if (kb + 1 < nk) {
      // Buffer cur^1 was last read at iteration kb-1, whose trailing barrier
      // precedes this issue -> no WAR hazard on LDS.
      issue_async(p, ib, jb, (kb + 1) << 5, t, As[cur ^ 1], Bs[cur ^ 1]);
      // Async loads complete in order per wave: <=4 outstanding means the
      // 4 ops of stage kb have landed; the 4 just issued may stay in flight.
      asm volatile("s_wait_asynccnt 0x4" ::: "memory");
    } else {
      asm volatile("s_wait_asynccnt 0x0" ::: "memory");
    }
    __syncthreads();
    mma_step(As[cur], Bs[cur], wi, wj, lane, acc);
    __syncthreads();
  }

  // Epilogue
  int l15 = lane & 15, lh = lane >> 4;
  float fgv = 0.0f, lrv = 0.0f;
  if constexpr (EPI == EPI_WNEW) { fgv = p.fgate[0]; lrv = p.lrate[0]; }
#pragma unroll
  for (int it = 0; it < 2; ++it) {
    int gi0 = ib * 128 + wi * 32 + it * 16 + lh * 8;
#pragma unroll
    for (int jt = 0; jt < 4; ++jt) {
      int gj = jb * 128 + wj * 64 + jt * 16 + l15;
      float bj = p.bias ? p.bias[gj] : 0.0f;
#pragma unroll
      for (int v = 0; v < 8; ++v) {
        int gi = gi0 + v;
        if (gi >= p.I) continue;
        float val = acc[it][jt][v] + bj;
        if constexpr (EPI == EPI_SILU) {
          val = val / (1.0f + __expf(-val));
        }
        if constexpr (EPI == EPI_DIFF) {
          val -= p.aux[(size_t)gi * p.J + gj];
          if (gi >= p.Nvalid) val = 0.0f;
        }
        if constexpr (EPI == EPI_WNEW) {
          val = (1.0f - fgv) * p.aux[(size_t)gi * p.J + gj] +
                lrv * (p.scale * val);
        }
        if constexpr (EPI == EPI_MO) {
          int bb = (gi >= (S_LEN + SLOTS)) ? 1 : 0;
          int ss = gi - bb * (S_LEN + SLOTS);
          if (ss < S_LEN) {
            int tt = bb * S_LEN + ss;
            p.moF[(size_t)tt * H_DIM + gj] = val;
            p.agate[(size_t)tt * 2 * H_DIM + H_DIM + gj] = (bf16)val;
          }
        } else {
          if (p.outF) p.outF[(size_t)gi * p.J + gj] = val;
          if (p.outB) p.outB[(size_t)gi * p.J + gj] = (bf16)val;
        }
      }
    }
  }
}

// ---------------------------------------------------------------------------
// Elementwise / reduction kernels (fp32, bandwidth-trivial)
// ---------------------------------------------------------------------------

static __device__ __forceinline__ float block_reduce_sum(float v, float* red) {
  int t = threadIdx.x;
  red[t] = v;
  __syncthreads();
  for (int s = 128; s > 0; s >>= 1) {
    if (t < s) red[t] += red[t + s];
    __syncthreads();
  }
  float r = red[0];
  __syncthreads();
  return r;
}

// Sliding-window attention (17-wide) + residual + LayerNorm1.
// One block (256 thr) per token. Writes x1 f32 and the LEFT half of agate.
__global__ __launch_bounds__(256) void attn_ln1_kernel(
    const float* __restrict__ x, const float* __restrict__ g1,
    const float* __restrict__ be1, float* __restrict__ x1f,
    bf16* __restrict__ agate) {
  int tok = blockIdx.x;
  int b = tok >> 12;
  int s = tok & (S_LEN - 1);
  int t = threadIdx.x;
  __shared__ float xs[H_DIM];
  __shared__ float sc[32];
  __shared__ float red[256];

  const float* xrow = x + (size_t)tok * H_DIM;
  float4 xv = ((const float4*)xrow)[t];
  ((float4*)xs)[t] = xv;
  __syncthreads();

  int wave = t >> 5, lane = t & 31;
  for (int w = wave; w < 17; w += 8) {
    int j = s - 16 + w;
    float partial = 0.0f;
    if (j >= 0) {
      const float* krow = x + ((size_t)b * S_LEN + j) * H_DIM;
      for (int h = lane; h < H_DIM; h += 32) partial += xs[h] * krow[h];
    }
#pragma unroll
    for (int off = 16; off > 0; off >>= 1)
      partial += __shfl_xor(partial, off, 32);
    if (lane == 0) sc[w] = (j >= 0) ? partial * 0.03125f : -1e30f;
  }
  __syncthreads();

  float mx = -1e30f;
#pragma unroll
  for (int w = 0; w < 17; ++w) mx = fmaxf(mx, sc[w]);
  float pr[17];
  float den = 0.0f;
#pragma unroll
  for (int w = 0; w < 17; ++w) { pr[w] = __expf(sc[w] - mx); den += pr[w]; }
  float inv = 1.0f / den;

  float4 ao = {0.f, 0.f, 0.f, 0.f};
#pragma unroll
  for (int w = 0; w < 17; ++w) {
    int j = s - 16 + w;
    if (j < 0) continue;
    const float4* krow = (const float4*)(x + ((size_t)b * S_LEN + j) * H_DIM);
    float4 kv = krow[t];
    float pw = pr[w] * inv;
    ao.x += pw * kv.x; ao.y += pw * kv.y; ao.z += pw * kv.z; ao.w += pw * kv.w;
  }
  float4 y;
  y.x = xv.x + ao.x; y.y = xv.y + ao.y; y.z = xv.z + ao.z; y.w = xv.w + ao.w;

  float sum = block_reduce_sum(y.x + y.y + y.z + y.w, red);
  float mean = sum * (1.0f / H_DIM);
  float d0 = y.x - mean, d1 = y.y - mean, d2 = y.z - mean, d3 = y.w - mean;
  float ss2 = block_reduce_sum(d0 * d0 + d1 * d1 + d2 * d2 + d3 * d3, red);
  float rin = rsqrtf(ss2 * (1.0f / H_DIM) + 1e-5f);
  int h0 = t * 4;
  float yv[4] = {d0, d1, d2, d3};
#pragma unroll
  for (int k = 0; k < 4; ++k) {
    float hv = yv[k] * rin * g1[h0 + k] + be1[h0 + k];
    x1f[(size_t)tok * H_DIM + h0 + k] = hv;
    agate[(size_t)tok * 2 * H_DIM + h0 + k] = (bf16)hv;
  }
}

// Row-normalize kv rows (x then P per batch), zero-pad to NPAD, emit bf16.
__global__ __launch_bounds__(256) void kn_kernel(const float* __restrict__ x,
                                                 const float* __restrict__ P,
                                                 bf16* __restrict__ kn) {
  int n = blockIdx.x;  // 0..NPAD-1
  int t = threadIdx.x;
  __shared__ float red[256];
  const float* src = nullptr;
  if (n < NKV) {
    int b = (n < (S_LEN + SLOTS)) ? 0 : 1;
    int r = n - b * (S_LEN + SLOTS);
    src = (r < S_LEN) ? x + ((size_t)b * S_LEN + r) * H_DIM
                      : P + (size_t)(r - S_LEN) * H_DIM;
  }
  float4 v = {0.f, 0.f, 0.f, 0.f};
  if (src) v = ((const float4*)src)[t];
  float tot = block_reduce_sum(v.x * v.x + v.y * v.y + v.z * v.z + v.w * v.w, red);
  float scl = src ? (1.0f / fmaxf(sqrtf(tot), 1e-12f)) : 0.0f;
  bf16* o = kn + (size_t)n * H_DIM + t * 4;
  o[0] = (bf16)(v.x * scl); o[1] = (bf16)(v.y * scl);
  o[2] = (bf16)(v.z * scl); o[3] = (bf16)(v.w * scl);
}

// gate = sigmoid(glog); combined = g*x1+(1-g)*mo; x2 = LN(x1+combined)
__global__ __launch_bounds__(256) void gate_ln2_kernel(
    const float* __restrict__ x1, const float* __restrict__ mo,
    const float* __restrict__ glog, const float* __restrict__ g2,
    const float* __restrict__ be2, float* __restrict__ x2f,
    bf16* __restrict__ x2b) {
  int tok = blockIdx.x;
  int t = threadIdx.x;
  __shared__ float red[256];
  float4 a = ((const float4*)(x1 + (size_t)tok * H_DIM))[t];
  float4 m = ((const float4*)(mo + (size_t)tok * H_DIM))[t];
  float4 l = ((const float4*)(glog + (size_t)tok * H_DIM))[t];
  float av[4] = {a.x, a.y, a.z, a.w};
  float mv[4] = {m.x, m.y, m.z, m.w};
  float lv[4] = {l.x, l.y, l.z, l.w};
  float yv[4];
#pragma unroll
  for (int k = 0; k < 4; ++k) {
    float g = 1.0f / (1.0f + __expf(-lv[k]));
    float comb = g * av[k] + (1.0f - g) * mv[k];
    yv[k] = av[k] + comb;
  }
  float sum = block_reduce_sum(yv[0] + yv[1] + yv[2] + yv[3], red);
  float mean = sum * (1.0f / H_DIM);
  float ssl = 0.0f;
#pragma unroll
  for (int k = 0; k < 4; ++k) { yv[k] -= mean; ssl += yv[k] * yv[k]; }
  float ss2 = block_reduce_sum(ssl, red);
  float rin = rsqrtf(ss2 * (1.0f / H_DIM) + 1e-5f);
  int h0 = t * 4;
#pragma unroll
  for (int k = 0; k < 4; ++k) {
    float hv = yv[k] * rin * g2[h0 + k] + be2[h0 + k];
    x2f[(size_t)tok * H_DIM + h0 + k] = hv;
    x2b[(size_t)tok * H_DIM + h0 + k] = (bf16)hv;
  }
}

// out = LN(x2 + ff)
__global__ __launch_bounds__(256) void ln3_kernel(
    const float* __restrict__ x2, const float* __restrict__ ff,
    const float* __restrict__ g3, const float* __restrict__ be3,
    float* __restrict__ out) {
  int tok = blockIdx.x;
  int t = threadIdx.x;
  __shared__ float red[256];
  float4 a = ((const float4*)(x2 + (size_t)tok * H_DIM))[t];
  float4 f = ((const float4*)(ff + (size_t)tok * H_DIM))[t];
  float yv[4] = {a.x + f.x, a.y + f.y, a.z + f.z, a.w + f.w};
  float sum = block_reduce_sum(yv[0] + yv[1] + yv[2] + yv[3], red);
  float mean = sum * (1.0f / H_DIM);
  float ssl = 0.0f;
#pragma unroll
  for (int k = 0; k < 4; ++k) { yv[k] -= mean; ssl += yv[k] * yv[k]; }
  float ss2 = block_reduce_sum(ssl, red);
  float rin = rsqrtf(ss2 * (1.0f / H_DIM) + 1e-5f);
  int h0 = t * 4;
#pragma unroll
  for (int k = 0; k < 4; ++k)
    out[(size_t)tok * H_DIM + h0 + k] = yv[k] * rin * g3[h0 + k] + be3[h0 + k];
}

// f32 -> bf16 (4 per thread)
__global__ __launch_bounds__(256) void f2bf4_kernel(const float* __restrict__ s,
                                                    bf16* __restrict__ d,
                                                    int n4) {
  int i = blockIdx.x * 256 + threadIdx.x;
  if (i < n4) {
    float4 v = ((const float4*)s)[i];
    d[4 * i + 0] = (bf16)v.x; d[4 * i + 1] = (bf16)v.y;
    d[4 * i + 2] = (bf16)v.z; d[4 * i + 3] = (bf16)v.w;
  }
}

// 32x32 LDS tile transpose, bf16.  src R x C -> dst C x R (R,C mult of 32)
__global__ __launch_bounds__(256) void transpose_bf16(
    const bf16* __restrict__ src, bf16* __restrict__ dst, int R, int C) {
  __shared__ bf16 tile[32][33];
  int tx = threadIdx.x & 31, ty = threadIdx.x >> 5;
  int c0 = blockIdx.x * 32, r0 = blockIdx.y * 32;
#pragma unroll
  for (int i = 0; i < 32; i += 8)
    tile[ty + i][tx] = src[(size_t)(r0 + ty + i) * C + c0 + tx];
  __syncthreads();
#pragma unroll
  for (int i = 0; i < 32; i += 8)
    dst[(size_t)(c0 + ty + i) * R + r0 + tx] = tile[tx][ty + i];
}

// ---------------------------------------------------------------------------
// Host-side orchestration
// ---------------------------------------------------------------------------

extern "C" void kernel_launch(void* const* d_in, const int* in_sizes, int n_in,
                              void* d_out, int out_size, void* d_ws,
                              size_t ws_size, hipStream_t stream) {
  (void)in_sizes; (void)n_in; (void)out_size; (void)ws_size;
  const float* x     = (const float*)d_in[0];
  const float* memW  = (const float*)d_in[1];
  const float* memb  = (const float*)d_in[2];
  const float* P     = (const float*)d_in[3];
  const float* Wi    = (const float*)d_in[4];
  const float* bi    = (const float*)d_in[5];
  const float* Wl1   = (const float*)d_in[6];
  const float* bl1   = (const float*)d_in[7];
  const float* Wl2   = (const float*)d_in[8];
  const float* bl2   = (const float*)d_in[9];
  const float* Wo    = (const float*)d_in[10];
  const float* bo    = (const float*)d_in[11];
  const float* fgate = (const float*)d_in[12];
  const float* lrate = (const float*)d_in[13];
  const float* Wg    = (const float*)d_in[14];
  const float* bg    = (const float*)d_in[15];
  const float* W1    = (const float*)d_in[16];
  const float* b1    = (const float*)d_in[17];
  const float* W2    = (const float*)d_in[18];
  const float* b2    = (const float*)d_in[19];
  const float* g1    = (const float*)d_in[20];
  const float* be1   = (const float*)d_in[21];
  const float* g2    = (const float*)d_in[22];
  const float* be2   = (const float*)d_in[23];
  const float* g3    = (const float*)d_in[24];
  const float* be3   = (const float*)d_in[25];
  float* out = (float*)d_out;

  uintptr_t base = (uintptr_t)d_ws;
  size_t off = 0;
  auto alloc = [&](size_t bytes) -> void* {
    void* p = (void*)(base + off);
    off = off + ((bytes + 255) & ~(size_t)255);
    return p;
  };

  bf16* wWi   = (bf16*)alloc((size_t)H_DIM * H_DIM * 2);
  bf16* wWl1  = (bf16*)alloc((size_t)H_DIM * H_DIM * 2);
  bf16* wWl2  = (bf16*)alloc((size_t)H_DIM * H_DIM * 2);
  bf16* wMemW = (bf16*)alloc((size_t)H_DIM * H_DIM * 2);
  bf16* wWo   = (bf16*)alloc((size_t)H_DIM * H_DIM * 2);
  bf16* wWg   = (bf16*)alloc((size_t)H_DIM * 2 * H_DIM * 2);
  bf16* wW1   = (bf16*)alloc((size_t)FF_DIM * H_DIM * 2);
  bf16* wW2   = (bf16*)alloc((size_t)H_DIM * FF_DIM * 2);
  float* x1f  = (float*)alloc((size_t)TOK * H_DIM * 4);
  bf16* knB   = (bf16*)alloc((size_t)NPAD * H_DIM * 2);
  float* kpF  = (float*)alloc((size_t)NPAD * H_DIM * 4);
  bf16* kpB   = (bf16*)alloc((size_t)NPAD * H_DIM * 2);
  bf16* h1B   = (bf16*)alloc((size_t)NPAD * H_DIM * 2);
  bf16* hddB  = (bf16*)alloc((size_t)NPAD * H_DIM * 2);
  bf16* diffB = (bf16*)alloc((size_t)NPAD * H_DIM * 2);
  bf16* diffT = (bf16*)alloc((size_t)H_DIM * NPAD * 2);
  bf16* hddT  = (bf16*)alloc((size_t)H_DIM * NPAD * 2);
  bf16* wnewB = (bf16*)alloc((size_t)H_DIM * H_DIM * 2);
  bf16* moutB = (bf16*)alloc((size_t)NPAD * H_DIM * 2);
  float* moF  = (float*)alloc((size_t)TOK * H_DIM * 4);
  bf16* agate = (bf16*)alloc((size_t)TOK * 2 * H_DIM * 2);
  float* glog = (float*)alloc((size_t)TOK * H_DIM * 4);
  float* x2f  = (float*)alloc((size_t)TOK * H_DIM * 4);
  bf16* x2b   = (bf16*)alloc((size_t)TOK * H_DIM * 2);
  bf16* hffB  = (bf16*)alloc((size_t)TOK * FF_DIM * 2);
  float* ffF  = (float*)alloc((size_t)TOK * H_DIM * 4);

  auto cvt = [&](const float* s, bf16* d, size_t n) {
    int n4 = (int)(n / 4);
    f2bf4_kernel<<<(n4 + 255) / 256, 256, 0, stream>>>(s, d, n4);
  };
  cvt(Wi,   wWi,   (size_t)H_DIM * H_DIM);
  cvt(Wl1,  wWl1,  (size_t)H_DIM * H_DIM);
  cvt(Wl2,  wWl2,  (size_t)H_DIM * H_DIM);
  cvt(memW, wMemW, (size_t)H_DIM * H_DIM);
  cvt(Wo,   wWo,   (size_t)H_DIM * H_DIM);
  cvt(Wg,   wWg,   (size_t)H_DIM * 2 * H_DIM);
  cvt(W1,   wW1,   (size_t)FF_DIM * H_DIM);
  cvt(W2,   wW2,   (size_t)H_DIM * FF_DIM);

  attn_ln1_kernel<<<TOK, 256, 0, stream>>>(x, g1, be1, x1f, agate);
  kn_kernel<<<NPAD, 256, 0, stream>>>(x, P, knB);

  auto gemm = [&](int epi, const bf16* A, const bf16* B, const float* bias,
                  float* oF, bf16* oB, const float* aux, float scale, int I,
                  int J, int K, int Nvalid, float* mo_, bf16* ag_) {
    GemmP p{A, B, bias, oF, oB, aux, fgate, lrate, scale,
            I, J, K, Nvalid, mo_, ag_};
    dim3 g(J / 128, (I + 127) / 128);
    switch (epi) {
      case EPI_NONE: gemm_k<EPI_NONE><<<g, 256, 0, stream>>>(p); break;
      case EPI_SILU: gemm_k<EPI_SILU><<<g, 256, 0, stream>>>(p); break;
      case EPI_DIFF: gemm_k<EPI_DIFF><<<g, 256, 0, stream>>>(p); break;
      case EPI_WNEW: gemm_k<EPI_WNEW><<<g, 256, 0, stream>>>(p); break;
      case EPI_MO:   gemm_k<EPI_MO>  <<<g, 256, 0, stream>>>(p); break;
    }
  };

  // kp = kn @ Wi^T + bi   (keep f32 for diff, bf16 for next GEMM)
  gemm(EPI_NONE, knB, wWi, bi, kpF, kpB, nullptr, 0.f,
       NPAD, H_DIM, H_DIM, NPAD, nullptr, nullptr);
  // h1 = silu(kp @ Wl1^T + bl1)
  gemm(EPI_SILU, kpB, wWl1, bl1, nullptr, h1B, nullptr, 0.f,
       NPAD, H_DIM, H_DIM, NPAD, nullptr, nullptr);
  // hdd = silu(h1 @ Wl2^T + bl2)
  gemm(EPI_SILU, h1B, wWl2, bl2, nullptr, hddB, nullptr, 0.f,
       NPAD, H_DIM, H_DIM, NPAD, nullptr, nullptr);
  // diff = (hdd @ mem_W^T + mem_b) - kp  (zero padded rows)
  gemm(EPI_DIFF, hddB, wMemW, memb, nullptr, diffB, kpF, 0.f,
       NPAD, H_DIM, H_DIM, NKV, nullptr, nullptr);
  // transposes for grad GEMM
  {
    dim3 g(H_DIM / 32, NPAD / 32);
    transpose_bf16<<<g, 256, 0, stream>>>(diffB, diffT, NPAD, H_DIM);
    transpose_bf16<<<g, 256, 0, stream>>>(hddB, hddT, NPAD, H_DIM);
  }
  // W_new = (1-fgate)*mem_W + lrate * (2/(n*m)) * diff^T @ hdd
  gemm(EPI_WNEW, diffT, hddT, nullptr, nullptr, wnewB, memW,
       2.0f / ((float)NKV * (float)H_DIM),
       H_DIM, H_DIM, NPAD, H_DIM, nullptr, nullptr);
  // mem_out = hdd @ W_new^T + mem_b
  gemm(EPI_NONE, hddB, wnewB, memb, nullptr, moutB, nullptr, 0.f,
       NPAD, H_DIM, H_DIM, NPAD, nullptr, nullptr);
  // mo = mem_out @ Wo^T + bo  (scatter valid rows into moF + agate right half)
  gemm(EPI_MO, moutB, wWo, bo, nullptr, nullptr, nullptr, 0.f,
       NPAD, H_DIM, H_DIM, NPAD, moF, agate);
  // gate logits = [x1, mo] @ Wg^T + bg
  gemm(EPI_NONE, agate, wWg, bg, glog, nullptr, nullptr, 0.f,
       TOK, H_DIM, 2 * H_DIM, TOK, nullptr, nullptr);

  gate_ln2_kernel<<<TOK, 256, 0, stream>>>(x1f, moF, glog, g2, be2, x2f, x2b);

  // ff = silu(x2 @ W1^T + b1) @ W2^T + b2
  gemm(EPI_SILU, x2b, wW1, b1, nullptr, hffB, nullptr, 0.f,
       TOK, FF_DIM, H_DIM, TOK, nullptr, nullptr);
  gemm(EPI_NONE, hffB, wW2, b2, ffF, nullptr, nullptr, 0.f,
       TOK, H_DIM, FF_DIM, TOK, nullptr, nullptr);

  ln3_kernel<<<TOK, 256, 0, stream>>>(x2f, ffF, g3, be3, out);
}
